// myLSTM_44942537785519
// MI455X (gfx1250) — compile-verified, matrix operand order checked by reading, not run on hardware
//
#include <hip/hip_runtime.h>
#include <math.h>

#define INPUT_DIM  256
#define HIDDEN_DIM 1024
#define GATES_DIM  4096   // 4*HIDDEN_DIM
#define BATCH      128
#define SEQ_LEN    1024
#define OUT_DIM    128
#define T_STEPS    (SEQ_LEN - 1)   // 1023

typedef __attribute__((ext_vector_type(16))) __bf16 v16bf;
typedef __attribute__((ext_vector_type(8)))  float  v8f;

#define WMMA_BF16(A, B, C) \
    __builtin_amdgcn_wmma_f32_16x16x32_bf16(false, (A), false, (B), (short)0, (C), false, false)

// sched_group_barrier masks
#define SG_VMEM_READ 0x020
#define SG_MATRIX    0x008

// ---------------------------------------------------------------------------
// Fragment loader for V_WMMA_F32_16X16X32_BF16.
// ISA layout (16-bit A 16x32): lanes 0-15 hold row M=lane, K = {0..7, 16..23};
// lanes 16-31 hold row M=lane-16, K = {8..15, 24..31}. B (32x16, stored as
// W[N][K] row-major) mirrors this with N in place of M. Both reduce to two
// contiguous 16-byte loads per lane from a row-major bf16 matrix.
// ---------------------------------------------------------------------------
__device__ __forceinline__ v16bf ld_frag_bf16(const __bf16* rowp, int k0, int lane) {
    const __bf16* p = rowp + k0 + ((lane >> 4) << 3);
    union { v16bf v; uint4 u[2]; } r;
    r.u[0] = *(const uint4*)(p);
    r.u[1] = *(const uint4*)(p + 16);
    return r.v;
}

__device__ __forceinline__ float sigmoidf_(float x) {
    return 1.0f / (1.0f + __expf(-x));
}

// ---------------------------------------------------------------------------
// Prep kernels
// ---------------------------------------------------------------------------
__global__ void cvt_f32_to_bf16(const float* __restrict__ src,
                                __bf16* __restrict__ dst, int n) {
    int i = blockIdx.x * blockDim.x + threadIdx.x;
    if (i < n) dst[i] = (__bf16)src[i];
}

__global__ void bias_sum_kernel(const float* __restrict__ b_ih,
                                const float* __restrict__ b_hh,
                                float* __restrict__ bias, int n) {
    int i = blockIdx.x * blockDim.x + threadIdx.x;
    if (i < n) bias[i] = b_ih[i] + b_hh[i];
}

__global__ void zero_state_kernel(__bf16* __restrict__ h0,
                                  float* __restrict__ c, int n) {
    int i = blockIdx.x * blockDim.x + threadIdx.x;
    if (i < n) { h0[i] = (__bf16)0.0f; c[i] = 0.0f; }
}

// ---------------------------------------------------------------------------
// One LSTM step. grid = (8, 4): x = group of 8 hidden tiles (128 h-cols),
// y = pair of batch tiles (32 rows). block = 256 = 8 waves; each wave owns a
// 32x16 output tile (2 M-tiles x 1 hidden tile x 4 gates = 8 accumulators),
// sharing the 4 gate B-fragments across both M-tiles.
// Two-deep software pipeline, enforced with sched_group_barrier: per K-step,
// 12 VMEM reads (fragments for k+1) are scheduled before the 8 WMMAs of k,
// so waits become loadcnt<=12 instead of 0 and fetches overlap the matrix
// pipe. h is double buffered (h_in read-only, h_out write-only).
// ---------------------------------------------------------------------------
__global__ void __launch_bounds__(256)
lstm_step_kernel(const __bf16* __restrict__ xseq,    // [B][SEQ][D] bf16
                 const __bf16* __restrict__ w_ih,    // [4H][D] bf16
                 const __bf16* __restrict__ w_hh,    // [4H][H] bf16
                 const float* __restrict__ bias,     // [4H] = b_ih+b_hh
                 const __bf16* __restrict__ h_in,    // [B][H]
                 __bf16* __restrict__ h_out,         // [B][H]
                 float* __restrict__ c,              // [B][H]
                 int t) {
    const int lane = threadIdx.x & 31;
    const int wave = threadIdx.x >> 5;
    const int m0 = blockIdx.y * 32;                       // batch tile-pair base
    const int h0 = (blockIdx.x * 8 + wave) * 16;          // hidden tile base
    const int frow = lane & 15;

    // A-operand row pointers (batch rows m0+frow and m0+16+frow)
    const __bf16* xrow0 = xseq + ((size_t)(m0 + frow) * SEQ_LEN + t) * INPUT_DIM;
    const __bf16* xrow1 = xseq + ((size_t)(m0 + 16 + frow) * SEQ_LEN + t) * INPUT_DIM;
    const __bf16* hrow0 = h_in + (size_t)(m0 + frow) * HIDDEN_DIM;
    const __bf16* hrow1 = h_in + (size_t)(m0 + 16 + frow) * HIDDEN_DIM;

    // B-operand row pointers: gate g lives at rows g*H + h0 + frow of W
    const __bf16* bih[4];
    const __bf16* bhh[4];
#pragma unroll
    for (int g = 0; g < 4; ++g) {
        bih[g] = w_ih + (size_t)(g * HIDDEN_DIM + h0 + frow) * INPUT_DIM;
        bhh[g] = w_hh + (size_t)(g * HIDDEN_DIM + h0 + frow) * HIDDEN_DIM;
    }

    // 8 accumulators: acc[mi][g], initialized with the (n-dependent) bias
    v8f acc[2][4];
#pragma unroll
    for (int g = 0; g < 4; ++g) {
        float bv = bias[g * HIDDEN_DIM + h0 + frow];
#pragma unroll
        for (int j = 0; j < 8; ++j) { acc[0][g][j] = bv; acc[1][g][j] = bv; }
    }

    // ---- pipeline prologue: K-step 0 of the x part ----
    v16bf a_c0 = ld_frag_bf16(xrow0, 0, lane);
    v16bf a_c1 = ld_frag_bf16(xrow1, 0, lane);
    v16bf b_c[4];
#pragma unroll
    for (int g = 0; g < 4; ++g) b_c[g] = ld_frag_bf16(bih[g], 0, lane);

    // ---- x part: K = 256 (8 K-steps); steady state does 7 ----
#pragma unroll
    for (int kk = 0; kk < 7; ++kk) {
        const int kn = (kk + 1) * 32;
        v16bf a_n0 = ld_frag_bf16(xrow0, kn, lane);
        v16bf a_n1 = ld_frag_bf16(xrow1, kn, lane);
        v16bf b_n[4];
#pragma unroll
        for (int g = 0; g < 4; ++g) b_n[g] = ld_frag_bf16(bih[g], kn, lane);
#pragma unroll
        for (int g = 0; g < 4; ++g) {
            acc[0][g] = WMMA_BF16(a_c0, b_c[g], acc[0][g]);
            acc[1][g] = WMMA_BF16(a_c1, b_c[g], acc[1][g]);
        }
        __builtin_amdgcn_sched_group_barrier(SG_VMEM_READ, 12, 0);
        __builtin_amdgcn_sched_group_barrier(SG_MATRIX, 8, 0);
        a_c0 = a_n0; a_c1 = a_n1;
#pragma unroll
        for (int g = 0; g < 4; ++g) b_c[g] = b_n[g];
    }

    // ---- bridge: last x K-step overlaps first h fragment loads ----
    {
        v16bf a_n0 = ld_frag_bf16(hrow0, 0, lane);
        v16bf a_n1 = ld_frag_bf16(hrow1, 0, lane);
        v16bf b_n[4];
#pragma unroll
        for (int g = 0; g < 4; ++g) b_n[g] = ld_frag_bf16(bhh[g], 0, lane);
#pragma unroll
        for (int g = 0; g < 4; ++g) {
            acc[0][g] = WMMA_BF16(a_c0, b_c[g], acc[0][g]);
            acc[1][g] = WMMA_BF16(a_c1, b_c[g], acc[1][g]);
        }
        __builtin_amdgcn_sched_group_barrier(SG_VMEM_READ, 12, 0);
        __builtin_amdgcn_sched_group_barrier(SG_MATRIX, 8, 0);
        a_c0 = a_n0; a_c1 = a_n1;
#pragma unroll
        for (int g = 0; g < 4; ++g) b_c[g] = b_n[g];
    }

    // ---- h part: K = 1024 (32 K-steps); steady state does 31 ----
    for (int kk = 0; kk < 31; ++kk) {
        const int kn = (kk + 1) * 32;
        v16bf a_n0 = ld_frag_bf16(hrow0, kn, lane);
        v16bf a_n1 = ld_frag_bf16(hrow1, kn, lane);
        v16bf b_n[4];
#pragma unroll
        for (int g = 0; g < 4; ++g) b_n[g] = ld_frag_bf16(bhh[g], kn, lane);
#pragma unroll
        for (int g = 0; g < 4; ++g) {
            acc[0][g] = WMMA_BF16(a_c0, b_c[g], acc[0][g]);
            acc[1][g] = WMMA_BF16(a_c1, b_c[g], acc[1][g]);
        }
        __builtin_amdgcn_sched_group_barrier(SG_VMEM_READ, 12, 0);
        __builtin_amdgcn_sched_group_barrier(SG_MATRIX, 8, 0);
        a_c0 = a_n0; a_c1 = a_n1;
#pragma unroll
        for (int g = 0; g < 4; ++g) b_c[g] = b_n[g];
    }

    // ---- pipeline epilogue ----
#pragma unroll
    for (int g = 0; g < 4; ++g) {
        acc[0][g] = WMMA_BF16(a_c0, b_c[g], acc[0][g]);
        acc[1][g] = WMMA_BF16(a_c1, b_c[g], acc[1][g]);
    }

    // Gate nonlinearities + state update, register local.
    // C/D layout: VGPR j -> M = j (lanes 0-15) or j+8 (lanes 16-31); N = lane&15.
    const int mofs = (lane >> 4) << 3;
    const int hcol = h0 + frow;
#pragma unroll
    for (int mi = 0; mi < 2; ++mi) {
#pragma unroll
        for (int j = 0; j < 8; ++j) {
            int m = m0 + mi * 16 + mofs + j;
            size_t idx = (size_t)m * HIDDEN_DIM + hcol;
            float ig = sigmoidf_(acc[mi][0][j]);
            float fg = sigmoidf_(acc[mi][1][j]);
            float gg = tanhf(acc[mi][2][j]);
            float og = sigmoidf_(acc[mi][3][j]);
            float cn = fg * c[idx] + ig * gg;
            c[idx] = cn;
            h_out[idx] = (__bf16)(og * tanhf(cn));
        }
    }
}

// ---------------------------------------------------------------------------
// Per-step head: logits = h @ W_out^T + b_out, then row-wise log_softmax.
// grid = 8 (one batch tile per block), block = 256 = 8 waves covering all
// 128 output columns, so the softmax reduction stays inside the block.
// Same pipelining scheme (4 VMEM reads + 1 WMMA per scheduled group).
// ---------------------------------------------------------------------------
__global__ void __launch_bounds__(256)
head_kernel(const __bf16* __restrict__ h,        // [B][H]
            const __bf16* __restrict__ w_out,    // [V][H] bf16
            const float* __restrict__ b_out,     // [V]
            float* __restrict__ out,             // [B][T-1][V]
            int t) {
    __shared__ float ls[16][128];
    __shared__ float rowc[16];

    const int lane = threadIdx.x & 31;
    const int wave = threadIdx.x >> 5;
    const int m0 = blockIdx.x * 16;
    const int frow = lane & 15;
    const int n0 = wave * 16;

    const __bf16* hrow = h + (size_t)(m0 + frow) * HIDDEN_DIM;
    const __bf16* wrow = w_out + (size_t)(n0 + frow) * HIDDEN_DIM;

    v8f acc;
    {
        float bv = b_out[n0 + frow];
#pragma unroll
        for (int j = 0; j < 8; ++j) acc[j] = bv;
    }

    v16bf a_c = ld_frag_bf16(hrow, 0, lane);
    v16bf b_c = ld_frag_bf16(wrow, 0, lane);
    for (int kk = 0; kk < 31; ++kk) {
        const int kn = (kk + 1) * 32;
        v16bf a_n = ld_frag_bf16(hrow, kn, lane);
        v16bf b_n = ld_frag_bf16(wrow, kn, lane);
        acc = WMMA_BF16(a_c, b_c, acc);
        __builtin_amdgcn_sched_group_barrier(SG_VMEM_READ, 4, 0);
        __builtin_amdgcn_sched_group_barrier(SG_MATRIX, 1, 0);
        a_c = a_n; b_c = b_n;
    }
    acc = WMMA_BF16(a_c, b_c, acc);

    const int mofs = (lane >> 4) << 3;
#pragma unroll
    for (int j = 0; j < 8; ++j)
        ls[mofs + j][n0 + frow] = acc[j];
    __syncthreads();

    if (threadIdx.x < 16) {
        int r = threadIdx.x;
        float mx = -INFINITY;
#pragma unroll 4
        for (int v = 0; v < 128; ++v) mx = fmaxf(mx, ls[r][v]);
        float s = 0.0f;
#pragma unroll 4
        for (int v = 0; v < 128; ++v) s += __expf(ls[r][v] - mx);
        rowc[r] = mx + logf(s);
    }
    __syncthreads();

    for (int idx = threadIdx.x; idx < 16 * 128; idx += 256) {
        int r = idx >> 7;
        int v = idx & 127;
        int b = m0 + r;
        out[((size_t)b * T_STEPS + t) * OUT_DIM + v] = ls[r][v] - rowc[r];
    }
}

// ---------------------------------------------------------------------------
// Host side
// ---------------------------------------------------------------------------
extern "C" void kernel_launch(void* const* d_in, const int* in_sizes, int n_in,
                              void* d_out, int out_size, void* d_ws, size_t ws_size,
                              hipStream_t stream) {
    const float* seq   = (const float*)d_in[0];   // [128,1024,256]
    const float* W_ih  = (const float*)d_in[1];   // [4096,256]
    const float* W_hh  = (const float*)d_in[2];   // [4096,1024]
    const float* b_ih  = (const float*)d_in[3];   // [4096]
    const float* b_hh  = (const float*)d_in[4];   // [4096]
    const float* W_out = (const float*)d_in[5];   // [128,1024]
    const float* b_out = (const float*)d_in[6];   // [128]
    float* out = (float*)d_out;                   // [128,1023,128]

    // Workspace carve-out (all offsets 256B aligned); ~75.5 MB total
    char* ws = (char*)d_ws;
    size_t off = 0;
    auto carve = [&](size_t bytes) {
        void* p = ws + off;
        off += (bytes + 255) & ~(size_t)255;
        return p;
    };
    __bf16* x_bf     = (__bf16*)carve((size_t)BATCH * SEQ_LEN * INPUT_DIM * 2); // 64 MB
    __bf16* w_ih_bf  = (__bf16*)carve((size_t)GATES_DIM * INPUT_DIM * 2);       // 2 MB
    __bf16* w_hh_bf  = (__bf16*)carve((size_t)GATES_DIM * HIDDEN_DIM * 2);      // 8 MB
    __bf16* w_out_bf = (__bf16*)carve((size_t)OUT_DIM * HIDDEN_DIM * 2);        // 256 KB
    float*  bias     = (float*)carve((size_t)GATES_DIM * 4);                    // 16 KB
    __bf16* h_buf0   = (__bf16*)carve((size_t)BATCH * HIDDEN_DIM * 2);          // 256 KB
    __bf16* h_buf1   = (__bf16*)carve((size_t)BATCH * HIDDEN_DIM * 2);          // 256 KB
    float*  c_state  = (float*)carve((size_t)BATCH * HIDDEN_DIM * 4);           // 512 KB
    (void)ws_size; (void)in_sizes; (void)n_in; (void)out_size;

    // --- one-time conversions (deterministic, every call) ---
    {
        int n;
        n = BATCH * SEQ_LEN * INPUT_DIM;
        cvt_f32_to_bf16<<<n / 256, 256, 0, stream>>>(seq, x_bf, n);
        n = GATES_DIM * INPUT_DIM;
        cvt_f32_to_bf16<<<n / 256, 256, 0, stream>>>(W_ih, w_ih_bf, n);
        n = GATES_DIM * HIDDEN_DIM;
        cvt_f32_to_bf16<<<n / 256, 256, 0, stream>>>(W_hh, w_hh_bf, n);
        n = OUT_DIM * HIDDEN_DIM;
        cvt_f32_to_bf16<<<n / 256, 256, 0, stream>>>(W_out, w_out_bf, n);
        bias_sum_kernel<<<GATES_DIM / 256, 256, 0, stream>>>(b_ih, b_hh, bias, GATES_DIM);
        n = BATCH * HIDDEN_DIM;
        zero_state_kernel<<<n / 256, 256, 0, stream>>>(h_buf0, c_state, n);
    }

    // --- recurrence: 1023 steps, h double-buffered ---
    __bf16* h_bufs[2] = { h_buf0, h_buf1 };
    dim3 step_grid(8, 4);   // 8 hidden groups x 4 batch tile-pairs
    for (int t = 0; t < T_STEPS; ++t) {
        __bf16* h_in  = h_bufs[t & 1];
        __bf16* h_out = h_bufs[(t + 1) & 1];
        lstm_step_kernel<<<step_grid, 256, 0, stream>>>(
            x_bf, w_ih_bf, w_hh_bf, bias, h_in, h_out, c_state, t);
        head_kernel<<<8, 256, 0, stream>>>(h_out, w_out_bf, b_out, out, t);
    }
}
